// DEA_GNN_JK_14843406975373
// MI455X (gfx1250) — matrix operand order repaired and checked by reading, not compiled, not run on hardware
//
#include <hip/hip_runtime.h>
#include <hip/hip_bf16.h>
#include <stdint.h>

typedef __bf16 bf16;
typedef __attribute__((ext_vector_type(16))) __bf16 v16bf;
typedef __attribute__((ext_vector_type(8)))  __bf16 v8bf;
typedef __attribute__((ext_vector_type(8)))  float  v8f;

#define NNODES 50000
#define NEDGE  800000
#define NCAND  65536
#define DDIM   256

__device__ __forceinline__ bf16 f2bf(float f) { return (bf16)f; }

// ---------------- elementwise / setup kernels ----------------

__global__ void k_fill(float* __restrict__ p, int n) {
    int i = blockIdx.x * blockDim.x + threadIdx.x;
    if (i < n) p[i] = 0.0f;
}

__global__ void k_degree(const int* __restrict__ dst, float* __restrict__ deg, int e) {
    int i = blockIdx.x * blockDim.x + threadIdx.x;
    if (i < e) atomicAdd(&deg[dst[i]], 1.0f);
}

__global__ void k_dis(const float* __restrict__ deg, float* __restrict__ dis, int n) {
    int i = blockIdx.x * blockDim.x + threadIdx.x;
    if (i < n) {
        float d = deg[i];
        dis[i] = (d > 0.0f) ? rsqrtf(fmaxf(d, 1.0f)) : 0.0f;
    }
}

__global__ void k_concat(const float* __restrict__ emb, const float* __restrict__ feat,
                         float* __restrict__ x, int n) {
    int i = blockIdx.x * blockDim.x + threadIdx.x;
    if (i < n) {
        int node = i >> 8, c = i & 255;
        x[i] = (c < 128) ? emb[node * 128 + c] : feat[node * 128 + (c - 128)];
    }
}

__global__ void k_cvt_bf16(const float* __restrict__ in, bf16* __restrict__ out, int n) {
    int i = blockIdx.x * blockDim.x + threadIdx.x;
    if (i < n) out[i] = f2bf(in[i]);
}

// Wt[j*256+i] = (bf16) W[i*256+j]  (store W^T so B-fragment K runs are contiguous)
__global__ void k_transcvt(const float* __restrict__ W, bf16* __restrict__ Wt) {
    int t = blockIdx.x * blockDim.x + threadIdx.x;   // 65536 threads
    int j = t >> 8, i = t & 255;
    Wt[t] = f2bf(W[i * 256 + j]);
}

// hn[dst] += dis[src]*dis[dst] * h[src]; 32 threads per edge, 8 floats each
__global__ void k_scatter(const int* __restrict__ src, const int* __restrict__ dst,
                          const float* __restrict__ dis, const float* __restrict__ h,
                          float* __restrict__ hn, int e) {
    int t = blockIdx.x * blockDim.x + threadIdx.x;
    int ei = t >> 5;
    if (ei >= e) return;
    int part = t & 31;
    int s = src[ei], d = dst[ei];
    float w = dis[s] * dis[d];
    const float4* hv = (const float4*)(h + (size_t)s * DDIM) + part * 2;
    float4 a = hv[0], b = hv[1];
    float* o = hn + (size_t)d * DDIM + part * 8;
    atomicAdd(o + 0, w * a.x); atomicAdd(o + 1, w * a.y);
    atomicAdd(o + 2, w * a.z); atomicAdd(o + 3, w * a.w);
    atomicAdd(o + 4, w * b.x); atomicAdd(o + 5, w * b.y);
    atomicAdd(o + 6, w * b.z); atomicAdd(o + 7, w * b.w);
}

__global__ void k_relu_jk(const float* __restrict__ acc, float* __restrict__ jk,
                          float* __restrict__ x, int n, int first) {
    int i = blockIdx.x * blockDim.x + threadIdx.x;
    if (i < n) {
        float v = fmaxf(acc[i], 0.0f);
        jk[i] = first ? v : fmaxf(jk[i], v);
        x[i] = v;
    }
}

__global__ void k_edgeprod(const float* __restrict__ jk, const int* __restrict__ eli,
                           bf16* __restrict__ pbf, int n) {
    int i = blockIdx.x * blockDim.x + threadIdx.x;
    if (i < n) {
        int e = i >> 8, c = i & 255;
        int a = eli[e], b = eli[NCAND + e];
        pbf[i] = f2bf(jk[(size_t)a * DDIM + c] * jk[(size_t)b * DDIM + c]);
    }
}

// out[e] = h[e,:] . w1 + b1   (one wave32 per edge)
__global__ void k_dot(const float* __restrict__ h, const float* __restrict__ w1,
                      const float* __restrict__ b1, float* __restrict__ out) {
    int t = blockIdx.x * blockDim.x + threadIdx.x;
    int e = t >> 5, lane = t & 31;
    float s = 0.0f;
    const float* hp = h + (size_t)e * DDIM + lane * 8;
    const float* wp = w1 + lane * 8;
#pragma unroll
    for (int i = 0; i < 8; ++i) s += hp[i] * wp[i];
#pragma unroll
    for (int o = 16; o > 0; o >>= 1) s += __shfl_down(s, o, 32);
    if (lane == 0) out[e] = s + b1[0];
}

// ---------------- bf16 WMMA GEMM: C[M,256] (+)= A[M,256] @ Bt^T + bias ----------------
// Bt is the 256x256 weight stored transposed (Bt[n][k] = W[k][n]), bf16.
// Block = 256 threads = 8 waves; block covers 16 rows; wave w covers cols [32w,32w+32).
__global__ void __launch_bounds__(256) k_gemm_bf16(
    const bf16* __restrict__ A, const bf16* __restrict__ Bt,
    const float* __restrict__ bias, float* __restrict__ C,
    int accumulate, int relu) {
    const int lane = threadIdx.x & 31;
    const int wave = threadIdx.x >> 5;
    const int m    = lane & 15;
    const int hi   = lane >> 4;
    const int row0 = blockIdx.x * 16;
    const int n0   = wave * 32;

    const size_t arow = (size_t)(row0 + m) * DDIM;

    // C/D layout: element v -> row (v + 8*hi), col (lane & 15)
    v8f c0, c1;
    {
        float b0v = bias[n0 + m];
        float b1v = bias[n0 + 16 + m];
#pragma unroll
        for (int v = 0; v < 8; ++v) {
            size_t i0 = (size_t)(row0 + v + 8 * hi) * DDIM + n0 + m;
            c0[v] = b0v + (accumulate ? C[i0] : 0.0f);
            c1[v] = b1v + (accumulate ? C[i0 + 16] : 0.0f);
        }
    }

#pragma unroll
    for (int ks = 0; ks < 8; ++ks) {
        // A fragment 16x32: elems 0..15 = K 16*hi+0..15
        const bf16* abase = A + arow + ks * 32 + 16 * hi;
        v8bf alo = *(const v8bf*)(abase);
        v8bf ah8 = *(const v8bf*)(abase + 8);
        v16bf af;
#pragma unroll
        for (int i = 0; i < 8; ++i) { af[i] = alo[i]; af[8 + i] = ah8[i]; }

        // B fragment 32x16: col n = lane&15, elems 0..15 = K 16*hi+0..15 (contiguous in Bt)
        const bf16* bb0 = Bt + (size_t)(n0 + m) * DDIM + ks * 32 + 16 * hi;
        v16bf bfr0 = *(const v16bf*)bb0;
        v16bf bfr1 = *(const v16bf*)(bb0 + 16 * DDIM);

        c0 = __builtin_amdgcn_wmma_f32_16x16x32_bf16(false, af, false, bfr0,
                                                     (short)0, c0, false, false);
        c1 = __builtin_amdgcn_wmma_f32_16x16x32_bf16(false, af, false, bfr1,
                                                     (short)0, c1, false, false);
    }

#pragma unroll
    for (int v = 0; v < 8; ++v) {
        size_t i0 = (size_t)(row0 + v + 8 * hi) * DDIM + n0 + m;
        float v0 = c0[v], v1 = c1[v];
        if (relu) { v0 = fmaxf(v0, 0.0f); v1 = fmaxf(v1, 0.0f); }
        C[i0] = v0;
        C[i0 + 16] = v1;
    }
}

// ---------------- driver ----------------

extern "C" void kernel_launch(void* const* d_in, const int* in_sizes, int n_in,
                              void* d_out, int out_size, void* d_ws, size_t ws_size,
                              hipStream_t stream) {
    (void)in_sizes; (void)n_in; (void)out_size; (void)ws_size;

    const float* x_feature = (const float*)d_in[0];
    const float* emb       = (const float*)d_in[1];
    const float* W[3] = {(const float*)d_in[2], (const float*)d_in[4], (const float*)d_in[6]};
    const float* Bv[3] = {(const float*)d_in[3], (const float*)d_in[5], (const float*)d_in[7]};
    const float* mlp_w0 = (const float*)d_in[8];
    const float* mlp_b0 = (const float*)d_in[9];
    const float* mlp_w1 = (const float*)d_in[10];
    const float* mlp_b1 = (const float*)d_in[11];
    const int*   eidx   = (const int*)d_in[12];
    const int*   eli    = (const int*)d_in[13];
    const int*   esrc = eidx;
    const int*   edst = eidx + NEDGE;

    char* ws = (char*)d_ws;
    auto al = [](size_t s) { return (s + 255) & ~(size_t)255; };
    size_t off = 0;
    float* deg = (float*)(ws + off); off += al((size_t)NNODES * 4);
    float* dis = (float*)(ws + off); off += al((size_t)NNODES * 4);
    bf16*  Wt  = (bf16*)(ws + off);  off += al((size_t)13 * DDIM * DDIM * 2);
    bf16*  hbf = (bf16*)(ws + off);  off += al((size_t)NNODES * DDIM * 2);
    float* acc = (float*)(ws + off); off += al((size_t)NNODES * DDIM * 4);
    float* jk  = (float*)(ws + off); off += al((size_t)NNODES * DDIM * 4);
    size_t off_x = off;
    float* x   = (float*)(ws + off); off += al((size_t)NNODES * DDIM * 4);
    float* ha  = (float*)(ws + off); off += al((size_t)NNODES * DDIM * 4);
    size_t off_hb = off;
    float* hb  = (float*)(ws + off); off += al((size_t)NNODES * DDIM * 4);
    // MLP-phase reuse (layer buffers dead by then):
    bf16*  pbf  = (bf16*)(ws + off_hb);             // 33.5 MB <= 51.2 MB
    float* mlph = (float*)(ws + off_x);             // 67 MB spans x+ha (102.4 MB)

    const int T = 256;
    auto cdiv = [](int a, int b) { return (a + b - 1) / b; };
    const int nND = NNODES * DDIM;   // 12,800,000
    const int nED = NCAND * DDIM;    // 16,777,216

    // normalization
    k_fill  <<<cdiv(NNODES, T), T, 0, stream>>>(deg, NNODES);
    k_degree<<<cdiv(NEDGE, T), T, 0, stream>>>(edst, deg, NEDGE);
    k_dis   <<<cdiv(NNODES, T), T, 0, stream>>>(deg, dis, NNODES);

    // x0 = [emb | feat]
    k_concat<<<cdiv(nND, T), T, 0, stream>>>(emb, x_feature, x, nND);

    // weights -> bf16, transposed
    for (int l = 0; l < 3; ++l)
        for (int k = 0; k < 4; ++k)
            k_transcvt<<<256, T, 0, stream>>>(W[l] + (size_t)k * DDIM * DDIM,
                                              Wt + (size_t)(4 * l + k) * DDIM * DDIM);
    k_transcvt<<<256, T, 0, stream>>>(mlp_w0, Wt + (size_t)12 * DDIM * DDIM);

    // 3 TAGConv layers
    for (int l = 0; l < 3; ++l) {
        k_cvt_bf16<<<cdiv(nND, T), T, 0, stream>>>(x, hbf, nND);
        k_gemm_bf16<<<NNODES / 16, T, 0, stream>>>(hbf, Wt + (size_t)(4 * l) * DDIM * DDIM,
                                                   Bv[l], acc, 0, 0);
        const float* hcur = x;
        float* bufs[2] = {ha, hb};
        for (int k = 1; k <= 3; ++k) {
            float* hn = bufs[(k - 1) & 1];
            k_fill   <<<cdiv(nND, T), T, 0, stream>>>(hn, nND);
            k_scatter<<<cdiv(NEDGE * 32, T), T, 0, stream>>>(esrc, edst, dis, hcur, hn, NEDGE);
            k_cvt_bf16<<<cdiv(nND, T), T, 0, stream>>>(hn, hbf, nND);
            k_gemm_bf16<<<NNODES / 16, T, 0, stream>>>(hbf,
                                                       Wt + (size_t)(4 * l + k) * DDIM * DDIM,
                                                       Bv[l] + k * DDIM, acc, 1, 0);
            hcur = hn;
        }
        k_relu_jk<<<cdiv(nND, T), T, 0, stream>>>(acc, jk, x, nND, l == 0);
    }

    // edge scoring
    k_edgeprod<<<cdiv(nED, T), T, 0, stream>>>(jk, eli, pbf, nED);
    k_gemm_bf16<<<NCAND / 16, T, 0, stream>>>(pbf, Wt + (size_t)12 * DDIM * DDIM,
                                              mlp_b0, mlph, 0, 1);
    k_dot<<<cdiv(NCAND * 32, T), T, 0, stream>>>(mlph, mlp_w1, mlp_b1, (float*)d_out);
}
